// MoEBlock_24043226923898
// MI455X (gfx1250) — compile-verified, matrix operand order checked by reading, not run on hardware
//
#include <hip/hip_runtime.h>
#include <hip/hip_bf16.h>
#include <math.h>

// ---------------- problem constants (from reference) ----------------
constexpr int TKN  = 2 * 2048;   // B*S tokens
constexpr int DM   = 1024;       // model dim
constexpr int HD   = 4096;       // hidden dim
constexpr int NEXP = 8;          // experts
constexpr int HC   = 128;        // H chunk processed per iteration

// ---------------- types ----------------
typedef __bf16 v16bf __attribute__((ext_vector_type(16)));
typedef float  v8f   __attribute__((ext_vector_type(8)));

union AFrag { v16bf v; unsigned int u[8]; };

// ---------------- helpers ----------------
static __device__ __forceinline__ unsigned short bf16_bits(float f) {
  unsigned int u = __builtin_bit_cast(unsigned int, f);
  u += 0x7FFFu + ((u >> 16) & 1u);          // round-to-nearest-even
  return (unsigned short)(u >> 16);
}
static __device__ __forceinline__ unsigned int pack_bf16x2(float lo, float hi) {
  return (unsigned int)bf16_bits(lo) | ((unsigned int)bf16_bits(hi) << 16);
}
// CDNA5 16-bit A/B fragment K-pair offset for VGPR j, lane-half h (ISA 7.12.2).
static __device__ __forceinline__ int kpair_off(int j, int half) {
  return (j < 4) ? (j * 2 + half * 8) : (16 + (j - 4) * 2 + half * 8);
}
static __device__ __forceinline__ float gelu_tanh(float v) {
  return 0.5f * v * (1.0f + tanhf(0.7978845608028654f * (v + 0.044715f * v * v * v)));
}
// Load a pre-packed 32x16 bf16 B-tile: lane L's 8 fragment dwords are the
// 32 contiguous bytes at tile + L*32 -> two global_load_b128 per lane.
static __device__ __forceinline__ v16bf load_b_frag_packed(const unsigned short* tile, int lane) {
  AFrag f;
  const uint4* p = (const uint4*)tile + lane * 2;
  uint4 a0 = p[0], a1 = p[1];
  f.u[0] = a0.x; f.u[1] = a0.y; f.u[2] = a0.z; f.u[3] = a0.w;
  f.u[4] = a1.x; f.u[5] = a1.y; f.u[6] = a1.z; f.u[7] = a1.w;
  return f.v;
}

// ---------------- kernel 1: router (one wave32 per token) ----------------
__global__ void router_kernel(const float* __restrict__ x, const float* __restrict__ Wg,
                              const float* __restrict__ bg,
                              int* __restrict__ topIdx, float* __restrict__ topW) {
  const int wv   = threadIdx.x >> 5;
  const int lane = threadIdx.x & 31;
  const int t    = blockIdx.x * (blockDim.x >> 5) + wv;
  if (t >= TKN) return;
  const float* xr = x + (size_t)t * DM;
  float acc[NEXP];
#pragma unroll
  for (int e = 0; e < NEXP; ++e) acc[e] = 0.0f;
  for (int d = lane; d < DM; d += 32) {
    const float xv = xr[d];
    const float* wr = Wg + (size_t)d * NEXP;
#pragma unroll
    for (int e = 0; e < NEXP; ++e) acc[e] = fmaf(xv, wr[e], acc[e]);
  }
#pragma unroll
  for (int e = 0; e < NEXP; ++e) {
#pragma unroll
    for (int off = 16; off >= 1; off >>= 1) acc[e] += __shfl_xor(acc[e], off, 32);
  }
  if (lane == 0) {
    float v[NEXP];
#pragma unroll
    for (int e = 0; e < NEXP; ++e) v[e] = acc[e] + bg[e];
    int bi = 0; float bv = v[0];
#pragma unroll
    for (int e = 1; e < NEXP; ++e) if (v[e] > bv) { bv = v[e]; bi = e; }
    int si = (bi == 0) ? 1 : 0; float sv = v[si];
#pragma unroll
    for (int e = 0; e < NEXP; ++e) if (e != bi && v[e] > sv) { sv = v[e]; si = e; }
    const float e1  = expf(sv - bv);
    const float inv = 1.0f / (1.0f + e1);
    topIdx[2 * t]     = bi;
    topIdx[2 * t + 1] = si;
    topW[2 * t]       = inv;
    topW[2 * t + 1]   = e1 * inv;
  }
}

// ---------------- kernel 2: deterministic per-expert list build ----------------
__global__ void build_lists_kernel(const int* __restrict__ topIdx,
                                   int* __restrict__ lists, int* __restrict__ counts) {
  const int e    = blockIdx.x;       // one wave32 per expert
  const int lane = threadIdx.x;
  int base = 0;
  for (int start = 0; start < TKN * 2; start += 32) {
    const int slot  = start + lane;
    const bool hit  = (topIdx[slot] == e);
    const unsigned int m     = (unsigned int)__ballot(hit);
    const unsigned int below = m & ((1u << lane) - 1u);
    if (hit) lists[e * TKN + base + __popc(below)] = slot;   // slot = token*2 + k
    base += __popc(m);
  }
  if (lane == 0) counts[e] = base;
}

// ---------------- kernel 3: residual init ----------------
__global__ void residual_init_kernel(const float* __restrict__ x, float* __restrict__ out, int n) {
  const int i = blockIdx.x * blockDim.x + threadIdx.x;
  if (i < n) out[i] = x[i];
}

// ---------------- kernel 4: weight pre-pack fp32 -> WMMA-native bf16 tiles ----------
// src: [NEXP, K, N] fp32.  dst: per expert, tiles ordered [nt][kb], each tile =
// 512 ushort = the exact 32-lane x 8-dword VGPR image of a 32(K) x 16(N) B-fragment.
__global__ __launch_bounds__(256)
void pack_weights_kernel(const float* __restrict__ src, unsigned short* __restrict__ dst,
                         int K, int N) {
  __shared__ unsigned short tls[8 * 512];       // 8 tiles staged per block
  const int kb  = blockIdx.x;                   // K/32 blocks
  const int nb  = blockIdx.y;                   // N/128 blocks
  const int e   = blockIdx.z;
  const int tid = threadIdx.x;
  const int koff = tid >> 3;                    // 0..31 (K row within tile)
  const int nt   = tid & 7;                     // n-tile within 128-col block

  // fragment coordinates for this K row
  const int half = (koff >> 3) & 1;
  const int j    = ((koff >> 4) & 1) * 4 + ((koff & 7) >> 1);
  const int hi   = koff & 1;

  const float* s = src + (size_t)e * K * N + (size_t)(kb * 32 + koff) * N + nb * 128 + nt * 16;
#pragma unroll
  for (int i = 0; i < 16; i += 4) {
    const float4 f = *(const float4*)(s + i);
    tls[nt * 512 + ((half * 16 + i + 0) * 8 + j) * 2 + hi] = bf16_bits(f.x);
    tls[nt * 512 + ((half * 16 + i + 1) * 8 + j) * 2 + hi] = bf16_bits(f.y);
    tls[nt * 512 + ((half * 16 + i + 2) * 8 + j) * 2 + hi] = bf16_bits(f.z);
    tls[nt * 512 + ((half * 16 + i + 3) * 8 + j) * 2 + hi] = bf16_bits(f.w);
  }
  __syncthreads();
  // write 8 tiles with coalesced b128 stores
  const int to = tid >> 5, lane = tid & 31;
  const int ntG = nb * 8 + to;
  unsigned short* dtile = dst + (((size_t)e * (N / 16) + ntG) * (K / 32) + kb) * 512;
  const uint4* sp = (const uint4*)&tls[to * 512] + lane * 2;
  uint4* dp = (uint4*)dtile + lane * 2;
  dp[0] = sp[0];
  dp[1] = sp[1];
}

// ---------------- kernel 5: sparse expert FFN with bf16 WMMA ----------------
template <bool PACKED>
__global__ __launch_bounds__(256, 1)
void moe_ffn_kernel(const float* __restrict__ x,
                    const float* __restrict__ W1, const float* __restrict__ b1,
                    const float* __restrict__ W2, const float* __restrict__ b2,
                    const unsigned short* __restrict__ W1p,
                    const unsigned short* __restrict__ W2p,
                    const int* __restrict__ lists, const int* __restrict__ counts,
                    const float* __restrict__ topW, float* __restrict__ out) {
  constexpr int XSTR = DM + 8;   // ushort stride (pad to dodge bank conflicts)
  constexpr int HSTR = HC + 8;
  __shared__ unsigned short x_lds[16 * XSTR];   // 16 x DM bf16 token tile (~33 KB)
  __shared__ unsigned short h_lds[16 * HSTR];   // 16 x HC bf16 hidden tile
  __shared__ int   s_tok[16];
  __shared__ float s_w[16];

  const int e       = blockIdx.y;
  const int cnt     = counts[e];
  const int rowBase = blockIdx.x * 16;
  if (rowBase >= cnt) return;                    // uniform across block

  const int tid  = threadIdx.x;
  const int lane = tid & 31;
  const int wv   = tid >> 5;                     // 8 waves
  const int half = lane >> 4;
  const int ln16 = lane & 15;

  if (tid < 16) {
    const int r    = rowBase + tid;
    const int slot = (r < cnt) ? lists[e * TKN + r] : -1;
    s_tok[tid] = (slot >= 0) ? (slot >> 1) : 0;  // token index
    s_w[tid]   = (slot >= 0) ? topW[slot] : 0.0f;
  }
  __syncthreads();

  // cooperative fp32 -> bf16 load of the 16-token x tile
  {
    const int r  = tid >> 4;            // 0..15 row
    const int c0 = (tid & 15) * 64;     // 64 floats per thread
    const float4* xr = (const float4*)(x + (size_t)s_tok[r] * DM);
    unsigned int* dst = (unsigned int*)&x_lds[r * XSTR + c0];
#pragma unroll
    for (int i = 0; i < 16; ++i) {
      const float4 f = xr[(c0 >> 2) + i];
      dst[2 * i]     = pack_bf16x2(f.x, f.y);
      dst[2 * i + 1] = pack_bf16x2(f.z, f.w);
    }
  }
  __syncthreads();

  const float* W1e = W1 + (size_t)e * DM * HD;
  const float* W2e = W2 + (size_t)e * HD * DM;

  // persistent output accumulators: wave owns D columns [wv*128, wv*128+128)
  v8f yacc[8];
#pragma unroll
  for (int nt = 0; nt < 8; ++nt) {
    const float bv = b2[e * DM + wv * 128 + nt * 16 + ln16];
#pragma unroll
    for (int i = 0; i < 8; ++i) yacc[nt][i] = bv;
  }

  for (int ch = 0; ch < HD / HC; ++ch) {
    const int hBase = ch * HC;

    // ---- phase 1: h[16 x HC] = gelu(xTile @ W1[:, hBase:hBase+HC] + b1) ----
    {
      const int hcol = hBase + wv * 16 + ln16;          // this lane's B column
      const unsigned short* W1tiles =
          PACKED ? W1p + ((size_t)e * (HD / 16) + (size_t)(ch * 8 + wv)) * (DM / 32) * 512
                 : nullptr;
      const float b1v = b1[e * HD + hcol];
      v8f hacc;
#pragma unroll
      for (int i = 0; i < 8; ++i) hacc[i] = b1v;
      for (int kb = 0; kb < DM / 32; ++kb) {
        AFrag a, b;
#pragma unroll
        for (int j = 0; j < 8; ++j)
          a.u[j] = *(const unsigned int*)&x_lds[ln16 * XSTR + kb * 32 + kpair_off(j, half)];
        if constexpr (PACKED) {
          b.v = load_b_frag_packed(W1tiles + (size_t)kb * 512, lane);
        } else {
#pragma unroll
          for (int j = 0; j < 8; ++j) {
            const int k = kb * 32 + kpair_off(j, half);
            b.u[j] = pack_bf16x2(W1e[(size_t)k * HD + hcol],
                                 W1e[(size_t)(k + 1) * HD + hcol]);
          }
        }
        hacc = __builtin_amdgcn_wmma_f32_16x16x32_bf16(
            false, a.v, false, b.v, (short)0, hacc, false, false);
      }
#pragma unroll
      for (int i = 0; i < 8; ++i) {
        const int m = i + half * 8;   // C layout: VGPR i -> row i / 8+i by lane half
        h_lds[m * HSTR + wv * 16 + ln16] = bf16_bits(gelu_tanh(hacc[i]));
      }
    }
    __syncthreads();

    // prefetch next H-chunk of W1 tiles while phase 2 runs (global_prefetch_b8)
    if (ch + 1 < HD / HC) {
      if constexpr (PACKED) {
        __builtin_prefetch(
            W1p + ((size_t)e * (HD / 16) + (size_t)((ch + 1) * 8 + wv)) * (DM / 32) * 512 +
                (size_t)lane * 16, 0, 1);
      } else {
        __builtin_prefetch(&W1e[(size_t)lane * HD + hBase + HC], 0, 1);
      }
    }

    // ---- phase 2: yTile[:, wv slice] += h @ W2[hBase:hBase+HC, :] ----
    // A-fragments depend only on the K-step: hoist all 4 out of the N-tile loop.
    AFrag a2[4];
#pragma unroll
    for (int kkIdx = 0; kkIdx < 4; ++kkIdx)
#pragma unroll
      for (int j = 0; j < 8; ++j)
        a2[kkIdx].u[j] =
            *(const unsigned int*)&h_lds[ln16 * HSTR + kkIdx * 32 + kpair_off(j, half)];

#pragma unroll
    for (int nt = 0; nt < 8; ++nt) {
      const int ncol = wv * 128 + nt * 16 + ln16;       // output D column
      const unsigned short* W2tiles =
          PACKED ? W2p + (((size_t)e * (DM / 16) + (size_t)(wv * 8 + nt)) * (HD / 32) +
                          (size_t)(ch * 4)) * 512
                 : nullptr;
#pragma unroll
      for (int kkIdx = 0; kkIdx < 4; ++kkIdx) {
        AFrag b;
        if constexpr (PACKED) {
          b.v = load_b_frag_packed(W2tiles + (size_t)kkIdx * 512, lane);
        } else {
#pragma unroll
          for (int j = 0; j < 8; ++j) {
            const int k = kkIdx * 32 + kpair_off(j, half);
            b.u[j] = pack_bf16x2(W2e[(size_t)(hBase + k) * DM + ncol],
                                 W2e[(size_t)(hBase + k + 1) * DM + ncol]);
          }
        }
        yacc[nt] = __builtin_amdgcn_wmma_f32_16x16x32_bf16(
            false, a2[kkIdx].v, false, b.v, (short)0, yacc[nt], false, false);
      }
    }
    __syncthreads();   // h_lds rewritten next chunk
  }

  // ---- combine: out[token] += gate_w * y  (padding rows have gate_w == 0) ----
#pragma unroll
  for (int nt = 0; nt < 8; ++nt) {
    const int ncol = wv * 128 + nt * 16 + ln16;
#pragma unroll
    for (int i = 0; i < 8; ++i) {
      const int m = i + half * 8;
      atomicAdd(&out[(size_t)s_tok[m] * DM + ncol], s_w[m] * yacc[nt][i]);
    }
  }
}

// ---------------- host launcher ----------------
extern "C" void kernel_launch(void* const* d_in, const int* in_sizes, int n_in,
                              void* d_out, int out_size, void* d_ws, size_t ws_size,
                              hipStream_t stream) {
  (void)in_sizes; (void)n_in; (void)out_size;
  const float* x  = (const float*)d_in[0];
  const float* Wg = (const float*)d_in[1];
  const float* bg = (const float*)d_in[2];
  const float* W1 = (const float*)d_in[3];
  const float* b1 = (const float*)d_in[4];
  const float* W2 = (const float*)d_in[5];
  const float* b2 = (const float*)d_in[6];
  float* out = (float*)d_out;

  // workspace carve-out
  char* base = (char*)d_ws;
  size_t off = 0;
  int*   topIdx = (int*)(base + off);   off += (size_t)TKN * 2 * sizeof(int);
  float* topW   = (float*)(base + off); off += (size_t)TKN * 2 * sizeof(float);
  int*   counts = (int*)(base + off);   off += 256;
  int*   lists  = (int*)(base + off);   off += (size_t)NEXP * TKN * sizeof(int);
  off = (off + 1023) & ~(size_t)1023;
  unsigned short* W1p = (unsigned short*)(base + off);
  off += (size_t)NEXP * DM * HD * sizeof(unsigned short);
  unsigned short* W2p = (unsigned short*)(base + off);
  off += (size_t)NEXP * HD * DM * sizeof(unsigned short);
  const bool usePacked = (ws_size >= off);   // fixed per harness -> deterministic

  router_kernel<<<TKN / 4, 128, 0, stream>>>(x, Wg, bg, topIdx, topW);
  build_lists_kernel<<<NEXP, 32, 0, stream>>>(topIdx, lists, counts);
  residual_init_kernel<<<(TKN * DM + 255) / 256, 256, 0, stream>>>(x, out, TKN * DM);

  dim3 grid(TKN / 16, NEXP);   // worst case: every token routes to one expert
  if (usePacked) {
    pack_weights_kernel<<<dim3(DM / 32, HD / 128, NEXP), 256, 0, stream>>>(W1, W1p, DM, HD);
    pack_weights_kernel<<<dim3(HD / 32, DM / 128, NEXP), 256, 0, stream>>>(W2, W2p, HD, DM);
    moe_ffn_kernel<true><<<grid, 256, 0, stream>>>(x, W1, b1, W2, b2, W1p, W2p,
                                                   lists, counts, topW, out);
  } else {
    moe_ffn_kernel<false><<<grid, 256, 0, stream>>>(x, W1, b1, W2, b2, W1p, W2p,
                                                    lists, counts, topW, out);
  }
}
